// GeneratorGNN_65266323030111
// MI455X (gfx1250) — matrix operand order repaired and checked by reading, not compiled, not run on hardware
//
#include <hip/hip_runtime.h>

#define NN 50000
#define NE 200000
#define HID 128

typedef _Float16 h16_t;
typedef _Float16 v8h  __attribute__((ext_vector_type(8)));
typedef _Float16 v16h __attribute__((ext_vector_type(16)));
typedef float    v8f  __attribute__((ext_vector_type(8)));

union HV { v16h v; v8h h[2]; };

// ---------------- workspace layout (bytes) ----------------
static constexpr size_t OFF_W16  = 0;                                   // f16 transposed weights (<1MB)
static constexpr size_t OFF_XPAD = 1048576;                             // [N,32] f16 padded x
static constexpr size_t OFF_H32  = OFF_XPAD + (size_t)NN * 32 * 2;
static constexpr size_t OFF_H16  = OFF_H32  + (size_t)NN * HID * 4;
static constexpr size_t OFF_E32  = OFF_H16  + (size_t)NN * HID * 2;
static constexpr size_t OFF_E16  = OFF_E32  + (size_t)NE * HID * 4;
static constexpr size_t OFF_IN16 = OFF_E16  + (size_t)NE * HID * 2;     // [E,384] f16 layer-1 input
static constexpr size_t OFF_Y32  = OFF_IN16;                            // alias: raw edge features f32
static constexpr size_t OFF_ACTA = OFF_IN16 + (size_t)NE * 384 * 2;
static constexpr size_t OFF_ACTB = OFF_ACTA + (size_t)NE * HID * 2;
static constexpr size_t OFF_MSUM = OFF_ACTB + (size_t)NE * HID * 2;
static constexpr size_t OFF_MMAX = OFF_MSUM + (size_t)NN * HID * 4;
static constexpr size_t OFF_CNT  = OFF_MMAX + (size_t)NN * HID * 4;

// f16 W^T offsets (in halfs) inside OFF_W16
static constexpr size_t O_N0=0, O_N1=4096, O_N2=20480;
static constexpr size_t O_E0=36864, O_E1=69632, O_E2=86016;
static constexpr size_t O_F0=102400, O_F1=151552, O_F2=167936;
static constexpr size_t O_G0=184320, O_G1=233472, O_G2=249856;
static constexpr size_t O_FO0=266240, O_FO1=282624, O_FO2=299008;
static constexpr size_t O_S0=315392, O_S1=364544, O_S2=380928;

__device__ __forceinline__ unsigned fkey(float v) {
    unsigned b = __float_as_uint(v);
    return (b & 0x80000000u) ? ~b : (b | 0x80000000u);
}
__device__ __forceinline__ float fdecode(unsigned k) {
    unsigned b = (k & 0x80000000u) ? (k ^ 0x80000000u) : ~k;
    return __uint_as_float(b);
}

// ============ generic fused layer: Y = [LN+ReLU]( X(f16)[M,K] @ W^T(f16)[128,K] + b ) ============
// W^T is staged once per workgroup into LDS (<=96KB << 320KB WGP LDS); the K-loop then feeds
// WMMA B-operands from ds_load_b128 while VMEM streams activations (A) from HBM.
// optional: rawY32 = pre-residual linear output, out32 = res32 + Y, out16 = f16 copy of final.
__global__ __launch_bounds__(256) void gemm128_kernel(
    const h16_t* __restrict__ X, const h16_t* __restrict__ WT,
    const float* __restrict__ bias, const float* __restrict__ gamma,
    const float* __restrict__ beta, const float* __restrict__ res32,
    float* __restrict__ rawY32, float* __restrict__ out32,
    h16_t* __restrict__ out16, int M, int K)
{
    extern __shared__ __align__(16) h16_t sW[];          // [128][K] f16

    // cooperative weight stage (before any early exit so every wave reaches the barrier)
    {
        const int nchunk = K * 16;                       // 16-byte chunks: 128*K/8
        const v8h* src = (const v8h*)WT;
        v8h* dst = (v8h*)sW;
        for (int i = threadIdx.x; i < nchunk; i += 256) dst[i] = src[i];
    }
    __syncthreads();

    const int lane = threadIdx.x & 31;
    const int wave = threadIdx.x >> 5;
    const int rowbase = blockIdx.x * 128 + wave * 16;
    if (rowbase >= M) return;                 // M is a multiple of 16: wave-uniform exit
    const int l15 = lane & 15;
    const int hi  = lane >> 4;

    v8f c[8];
#pragma unroll
    for (int j = 0; j < 8; ++j) c[j] = v8f{};

    const h16_t* xrow = X + (size_t)(rowbase + l15) * (size_t)K;
    for (int k0 = 0; k0 < K; k0 += 32) {
        HV a;
        a.h[0] = *(const v8h*)(xrow + k0 + hi * 16);       // K = k0 + hi*16 .. +7
        a.h[1] = *(const v8h*)(xrow + k0 + hi * 16 + 8);   // K = k0+hi*16+8 .. +7
#pragma unroll
        for (int j = 0; j < 8; ++j) {
            const h16_t* wrow = sW + (size_t)(j * 16 + l15) * (size_t)K + k0 + hi * 16;
            HV b;
            b.h[0] = *(const v8h*)(wrow);
            b.h[1] = *(const v8h*)(wrow + 8);
            c[j] = __builtin_amdgcn_wmma_f32_16x16x32_f16(
                false, a.v, false, b.v, (short)0, c[j], false, false);
        }
    }

    float bv[8];
#pragma unroll
    for (int j = 0; j < 8; ++j) bv[j] = bias[j * 16 + l15];
#pragma unroll
    for (int j = 0; j < 8; ++j)
#pragma unroll
        for (int r = 0; r < 8; ++r) c[j][r] += bv[j];

    if (gamma) {                              // LayerNorm + ReLU
        float gv[8], tv[8];
#pragma unroll
        for (int j = 0; j < 8; ++j) { gv[j] = gamma[j*16 + l15]; tv[j] = beta[j*16 + l15]; }
#pragma unroll
        for (int r = 0; r < 8; ++r) {
            float s = 0.f, s2 = 0.f;
#pragma unroll
            for (int j = 0; j < 8; ++j) { float v = c[j][r]; s += v; s2 += v * v; }
            // row spread over 16 lanes of the same half-wave: xor-reduce stays in group
            for (int m = 1; m < 16; m <<= 1) {
                s  += __shfl_xor(s,  m, 32);
                s2 += __shfl_xor(s2, m, 32);
            }
            float mean = s * (1.0f / 128.0f);
            float var  = s2 * (1.0f / 128.0f) - mean * mean;
            float rstd = rsqrtf(var + 1e-5f);
#pragma unroll
            for (int j = 0; j < 8; ++j) {
                float v = (c[j][r] - mean) * rstd * gv[j] + tv[j];
                c[j][r] = v > 0.f ? v : 0.f;
            }
        }
    }

#pragma unroll
    for (int r = 0; r < 8; ++r) {
        size_t row = (size_t)(rowbase + hi * 8 + r);
#pragma unroll
        for (int j = 0; j < 8; ++j) {
            size_t idx = row * 128 + j * 16 + l15;
            float y = c[j][r];
            if (rawY32) rawY32[idx] = y;
            float f = res32 ? res32[idx] + y : y;
            if (out32) out32[idx] = f;
            if (out16) out16[idx] = (h16_t)f;
        }
    }
}

// ---------------- elementwise / helper kernels ----------------
__global__ void wt_convert_kernel(const float* __restrict__ W, h16_t* __restrict__ WT,
                                  int K, int N, int Kpad) {
    long long i = (long long)blockIdx.x * 256 + threadIdx.x;
    if (i >= (long long)N * Kpad) return;
    int n = (int)(i / Kpad), k = (int)(i - (long long)n * Kpad);
    WT[i] = (k < K) ? (h16_t)W[(size_t)k * N + n] : (h16_t)0.f;
}

__global__ void pad_x_kernel(const float* __restrict__ x, h16_t* __restrict__ xp) {
    long long i = (long long)blockIdx.x * 256 + threadIdx.x;
    if (i >= (long long)NN * 32) return;
    int node = (int)(i >> 5), c = (int)(i & 31);
    xp[i] = (c < 3) ? (h16_t)x[(size_t)node * 3 + c] : (h16_t)0.f;
}

__global__ void gather2_kernel(const h16_t* __restrict__ h, const int* __restrict__ ei,
                               h16_t* __restrict__ dst) {
    long long i = (long long)blockIdx.x * 256 + threadIdx.x;
    if (i >= (long long)NE * 256) return;
    int e = (int)(i >> 8), c = (int)(i & 255);
    int node = (c < 128) ? ei[e] : ei[NE + e];
    dst[i] = h[(size_t)node * 128 + (c & 127)];
}

__global__ void gather3_kernel(const h16_t* __restrict__ h, const h16_t* __restrict__ ef,
                               const int* __restrict__ ei, h16_t* __restrict__ dst) {
    long long i = (long long)blockIdx.x * 256 + threadIdx.x;
    if (i >= (long long)NE * 384) return;
    int e = (int)(i / 384), c = (int)(i - (long long)e * 384);
    h16_t v;
    if (c < 128)      v = h[(size_t)ei[e] * 128 + c];
    else if (c < 256) v = h[(size_t)ei[NE + e] * 128 + (c - 128)];
    else              v = ef[(size_t)e * 128 + (c - 256)];
    dst[i] = v;
}

__global__ void counts_kernel(const int* __restrict__ ei, float* __restrict__ cnt) {
    int i = blockIdx.x * 256 + threadIdx.x;
    if (i >= NE) return;
    atomicAdd(&cnt[ei[NE + i]], 1.0f);
}

__global__ void segment_kernel(const float* __restrict__ y, const int* __restrict__ ei,
                               float* __restrict__ msum, unsigned* __restrict__ mmax) {
    long long i = (long long)blockIdx.x * 256 + threadIdx.x;
    if (i >= (long long)NE * 128) return;
    int e = (int)(i >> 7), c = (int)(i & 127);
    int n = ei[NE + e];
    float v = y[i];
    atomicAdd(&msum[(size_t)n * 128 + c], v);
    atomicMax(&mmax[(size_t)n * 128 + c], fkey(v));
}

__global__ void buildnode_kernel(const h16_t* __restrict__ h, const float* __restrict__ msum,
                                 const unsigned* __restrict__ mmax, const float* __restrict__ cnt,
                                 h16_t* __restrict__ dst) {
    long long i = (long long)blockIdx.x * 256 + threadIdx.x;
    if (i >= (long long)NN * 384) return;
    int node = (int)(i / 384), c = (int)(i - (long long)node * 384);
    h16_t v;
    if (c < 128) v = h[(size_t)node * 128 + c];
    else if (c < 256) {
        float m = (cnt[node] > 0.f) ? fdecode(mmax[(size_t)node * 128 + (c - 128)]) : 0.f;
        v = (h16_t)m;
    } else v = (h16_t)msum[(size_t)node * 128 + (c - 256)];
    dst[i] = v;
}

__global__ void feat_out_l4_kernel(const h16_t* __restrict__ A, const float* __restrict__ W,
                                   const float* __restrict__ b, float* __restrict__ out) {
    long long i = (long long)blockIdx.x * 256 + threadIdx.x;
    if (i >= (long long)NN * 3) return;
    int node = (int)(i / 3), cc = (int)(i - (long long)node * 3);
    float s = b[cc];
    for (int k = 0; k < 128; ++k) s += (float)A[(size_t)node * 128 + k] * W[(size_t)k * 3 + cc];
    out[i] = s;
}

__global__ void scores_l4_kernel(const h16_t* __restrict__ A, const float* __restrict__ W,
                                 const float* __restrict__ b, float* __restrict__ out) {
    int i = blockIdx.x * 256 + threadIdx.x;
    if (i >= NE) return;
    float s = b[0];
    for (int k = 0; k < 128; ++k) s += (float)A[(size_t)i * 128 + k] * W[k];
    out[i] = 1.f / (1.f + expf(-s));
}

// ---------------- host orchestration ----------------
static inline unsigned gblk(long long n) { return (unsigned)((n + 255) / 256); }

extern "C" void kernel_launch(void* const* d_in, const int* in_sizes, int n_in,
                              void* d_out, int out_size, void* d_ws, size_t ws_size,
                              hipStream_t stream) {
    (void)in_sizes; (void)n_in; (void)out_size; (void)ws_size;
    char* wsb = (char*)d_ws;
    h16_t*   w16  = (h16_t*)(wsb + OFF_W16);
    h16_t*   xpad = (h16_t*)(wsb + OFF_XPAD);
    float*   h32  = (float*)(wsb + OFF_H32);
    h16_t*   h16p = (h16_t*)(wsb + OFF_H16);
    float*   e32  = (float*)(wsb + OFF_E32);
    h16_t*   e16p = (h16_t*)(wsb + OFF_E16);
    h16_t*   in16 = (h16_t*)(wsb + OFF_IN16);
    float*   y32  = (float*)(wsb + OFF_Y32);     // aliases in16 (disjoint lifetimes)
    h16_t*   acta = (h16_t*)(wsb + OFF_ACTA);
    h16_t*   actb = (h16_t*)(wsb + OFF_ACTB);
    float*   msum = (float*)(wsb + OFF_MSUM);
    unsigned* mmax = (unsigned*)(wsb + OFF_MMAX);
    float*   cnt  = (float*)(wsb + OFF_CNT);

    const float* x  = (const float*)d_in[0];
    const int*   ei = (const int*)d_in[1];
    auto F = [&](int i) { return (const float*)d_in[i]; };
    float* out = (float*)d_out;

    // ---- convert all 128-wide layer weights to f16 W^T ----
    struct WDesc { int idx; size_t off; int K, N, Kpad; };
    const WDesc wl[18] = {
        {3,  O_N0, 3,   128, 32 }, {7,  O_N1, 128, 128, 128}, {11, O_N2, 128, 128, 128},
        {13, O_E0, 256, 128, 256}, {17, O_E1, 128, 128, 128}, {21, O_E2, 128, 128, 128},
        {23, O_F0, 384, 128, 384}, {27, O_F1, 128, 128, 128}, {31, O_F2, 128, 128, 128},
        {33, O_G0, 384, 128, 384}, {37, O_G1, 128, 128, 128}, {41, O_G2, 128, 128, 128},
        {43, O_FO0,128, 128, 128}, {47, O_FO1,128, 128, 128}, {51, O_FO2,128, 128, 128},
        {57, O_S0, 384, 128, 384}, {61, O_S1, 128, 128, 128}, {65, O_S2, 128, 128, 128},
    };
    for (int i = 0; i < 18; ++i) {
        long long tot = (long long)wl[i].N * wl[i].Kpad;
        wt_convert_kernel<<<gblk(tot), 256, 0, stream>>>(F(wl[i].idx), w16 + wl[i].off,
                                                         wl[i].K, wl[i].N, wl[i].Kpad);
    }

    auto gemm = [&](const h16_t* X, size_t wtoff, int bidx, bool ln,
                    const float* res, float* raw, float* o32, h16_t* o16, int M, int K) {
        size_t ldsb = (size_t)128 * K * sizeof(h16_t);   // W^T staged in LDS
        gemm128_kernel<<<dim3((M + 127) / 128), 256, ldsb, stream>>>(
            X, w16 + wtoff, F(bidx),
            ln ? F(bidx + 1) : nullptr, ln ? F(bidx + 2) : nullptr,
            res, raw, o32, o16, M, K);
    };

    // degree counts (invariant over iterations)
    hipMemsetAsync(cnt, 0, (size_t)NN * 4, stream);
    counts_kernel<<<gblk(NE), 256, 0, stream>>>(ei, cnt);

    // ---- node encoder: pad x to K=32 then 3 WMMA layers ----
    pad_x_kernel<<<gblk((long long)NN * 32), 256, 0, stream>>>(x, xpad);
    gemm(xpad, O_N0, 4,  true,  nullptr, nullptr, nullptr, acta, NN, 32);
    gemm(acta, O_N1, 8,  true,  nullptr, nullptr, nullptr, actb, NN, 128);
    gemm(actb, O_N2, 12, false, nullptr, nullptr, h32,     h16p, NN, 128);

    // ---- edge encoder ----
    gather2_kernel<<<gblk((long long)NE * 256), 256, 0, stream>>>(h16p, ei, in16);
    gemm(in16, O_E0, 14, true,  nullptr, nullptr, nullptr, acta, NE, 256);
    gemm(acta, O_E1, 18, true,  nullptr, nullptr, nullptr, actb, NE, 128);
    gemm(actb, O_E2, 22, false, nullptr, nullptr, e32,     e16p, NE, 128);

    // ---- 4 message-passing iterations ----
    for (int it = 0; it < 4; ++it) {
        gather3_kernel<<<gblk((long long)NE * 384), 256, 0, stream>>>(h16p, e16p, ei, in16);
        gemm(in16, O_G0, 34, true,  nullptr, nullptr, nullptr, acta, NE, 384);
        gemm(acta, O_G1, 38, true,  nullptr, nullptr, nullptr, actb, NE, 128);
        // edge_features raw -> y32 ; e = e + edge_features (in place f32 + f16)
        gemm(actb, O_G2, 42, false, e32, y32, e32, e16p, NE, 128);

        hipMemsetAsync(msum, 0, (size_t)NN * 128 * 4 * 2, stream);  // msum + mmax contiguous
        segment_kernel<<<gblk((long long)NE * 128), 256, 0, stream>>>(y32, ei, msum, mmax);

        buildnode_kernel<<<gblk((long long)NN * 384), 256, 0, stream>>>(h16p, msum, mmax, cnt, in16);
        gemm(in16, O_F0, 24, true,  nullptr, nullptr, nullptr, acta, NN, 384);
        gemm(acta, O_F1, 28, true,  nullptr, nullptr, nullptr, actb, NN, 128);
        gemm(actb, O_F2, 32, false, h32, nullptr, h32, h16p, NN, 128);  // h = h + mlp(...)
    }

    // ---- edge scores head ----
    gather3_kernel<<<gblk((long long)NE * 384), 256, 0, stream>>>(h16p, e16p, ei, in16);
    gemm(in16, O_S0, 58, true, nullptr, nullptr, nullptr, acta, NE, 384);
    gemm(acta, O_S1, 62, true, nullptr, nullptr, nullptr, actb, NE, 128);
    gemm(actb, O_S2, 66, true, nullptr, nullptr, nullptr, acta, NE, 128);
    scores_l4_kernel<<<gblk(NE), 256, 0, stream>>>(acta, F(69), F(70), out + (size_t)NN * 3);

    // ---- node feature head ----
    gemm(h16p, O_FO0, 44, true, nullptr, nullptr, nullptr, actb, NN, 128);
    gemm(actb, O_FO1, 48, true, nullptr, nullptr, nullptr, acta, NN, 128);
    gemm(acta, O_FO2, 52, true, nullptr, nullptr, nullptr, actb, NN, 128);
    feat_out_l4_kernel<<<gblk((long long)NN * 3), 256, 0, stream>>>(actb, F(55), F(56), out);
}
